// GATConvCombined_13262859010606
// MI455X (gfx1250) — matrix operand order repaired and checked by reading, not compiled, not run on hardware
//
#include <hip/hip_runtime.h>
#include <hip/hip_bf16.h>

// ---------------------------------------------------------------------------
// Problem constants (match reference)
// ---------------------------------------------------------------------------
#define NN       50000     // nodes
#define NE       800000    // edges
#define FIN      128
#define NH       4         // heads
#define CC       64        // channels per head
#define HCDIM    256       // NH*CC
#define NG       64        // graphs
#define PDIM     128       // pool hidden
#define NEL      (NE + NN) // edges + self loops

typedef __attribute__((ext_vector_type(16))) __bf16 v16bf;
typedef __attribute__((ext_vector_type(8)))  float  v8f;

union BF16Frag {
  v16bf v;
  unsigned int u[8];
  uint4 q[2];
};

__device__ __forceinline__ unsigned int f2bf(float f) {
  unsigned int u = __float_as_uint(f);
  u += 0x7FFFu + ((u >> 16) & 1u);   // round-to-nearest-even
  return u >> 16;
}

__device__ __forceinline__ unsigned int pack2bf(float lo, float hi) {
  return f2bf(lo) | (f2bf(hi) << 16);
}

__device__ __forceinline__ void atomicMaxF(float* a, float v) {
  // monotonic bit-pattern trick (init value must be negative, e.g. -1e30)
  if (v >= 0.f) atomicMax(reinterpret_cast<int*>(a), __float_as_int(v));
  else          atomicMin(reinterpret_cast<unsigned int*>(a),
                          (unsigned int)__float_as_int(v));
}

// Build one lane's A fragment for k-slab kb (16-bit A 16x32 VGPR striping):
//   lanes 0-15  (khalf=0): VGPR p<4 -> K=2p,2p+1 ; p>=4 -> K=16+2(p-4),+1
//   lanes 16-31 (khalf=1): same +8
// Per lane the 16 values are two contiguous 8-float runs -> 4 x b128 loads.
__device__ __forceinline__ void load_a_frag(const float* __restrict__ A,
                                            int arow, bool rowOK, int K,
                                            int kb, int khalf, BF16Frag& f) {
  const float* Arow = A + (size_t)arow * K + kb * 32 + khalf * 8;
  float4 f0, f1, f2, f3;
  if (rowOK) {
    f0 = *reinterpret_cast<const float4*>(Arow + 0);
    f1 = *reinterpret_cast<const float4*>(Arow + 4);
    f2 = *reinterpret_cast<const float4*>(Arow + 16);
    f3 = *reinterpret_cast<const float4*>(Arow + 20);
  } else {
    f0 = f1 = f2 = f3 = make_float4(0.f, 0.f, 0.f, 0.f);
  }
  f.u[0] = pack2bf(f0.x, f0.y);
  f.u[1] = pack2bf(f0.z, f0.w);
  f.u[2] = pack2bf(f1.x, f1.y);
  f.u[3] = pack2bf(f1.z, f1.w);
  f.u[4] = pack2bf(f2.x, f2.y);
  f.u[5] = pack2bf(f2.z, f2.w);
  f.u[6] = pack2bf(f3.x, f3.y);
  f.u[7] = pack2bf(f3.z, f3.w);
}

// ---------------------------------------------------------------------------
// Fill kernel (ws is poisoned each run -> must init everything we accumulate)
// ---------------------------------------------------------------------------
__global__ void k_fill(float* __restrict__ p, float v, size_t n) {
  size_t i = (size_t)blockIdx.x * blockDim.x + threadIdx.x;
  if (i < n) p[i] = v;
}

// ---------------------------------------------------------------------------
// Convert fp32 weights (K x Nc, row major) into fragment-major bf16 layout
// matching the CDNA5 WMMA B-matrix VGPR striping:
//   frag(t, kb, lane): element i  <->  K = kb*32 + (lane>>4)*8 + (i<8 ? i : 8+i),
//                                      N = t*16 + (lane & 15)
// Stored contiguously: 32 bytes per (t,kb,lane) so the GEMM loads b128 x2.
// ---------------------------------------------------------------------------
__global__ void k_convert_w(const float* __restrict__ W,
                            unsigned short* __restrict__ out,
                            int K, int Nc) {
  int total = K * Nc;
  int idx = blockIdx.x * blockDim.x + threadIdx.x;
  if (idx >= total) return;
  int numKb = K >> 5;
  int i    = idx & 15;
  int lane = (idx >> 4) & 31;
  int r    = idx >> 9;            // r = t*numKb + kb
  int kb   = r % numKb;
  int t    = r / numKb;
  int col   = t * 16 + (lane & 15);
  int khalf = lane >> 4;
  int p = i >> 1, e = i & 1;
  int kk = kb * 32 + khalf * 8 + ((2 * p + e) < 8 ? (2 * p + e) : (2 * p + e) + 8);
  out[idx] = (unsigned short)f2bf(W[(size_t)kk * Nc + col]);
}

// ---------------------------------------------------------------------------
// WMMA GEMM: C[M,Nc] = A[M,K] (f32 -> bf16 on the fly) @ Bfrag, f32 accum.
// Block = 256 threads = 8 waves; wave w covers rows [blk.x*128 + w*16, +16)
// and 8 n-tiles (128 cols) selected by blockIdx.y.
// Per k-step: issue both 4-tile B load groups (clause-able b128s), run 4
// WMMAs on group 0 while group 1 is in flight, stage next A fragment (cvt
// VALU co-executes with the WMMA/XDL pipe), then 4 WMMAs on group 1.
// __launch_bounds__(256,1) lifts the VGPR budget so the pipelined schedule
// (~200 live VGPRs) fits without scratch spills.
// ---------------------------------------------------------------------------
__global__ void __launch_bounds__(256, 1)
k_gemm_bf16(const float* __restrict__ A,
            const unsigned short* __restrict__ Bfrag,
            float* __restrict__ C,
            int M, int K, int Nc) {
  const int lane  = threadIdx.x & 31;
  const int wave  = threadIdx.x >> 5;
  const int mBase = blockIdx.x * 128 + wave * 16;
  const int tBase = blockIdx.y * 8;
  const int numKb = K >> 5;
  const int khalf = lane >> 4;
  const int arow  = mBase + (lane & 15);
  const bool rowOK = (arow < M);

  v8f acc[8] = {};
  const uint4* Bq = reinterpret_cast<const uint4*>(Bfrag);

  BF16Frag af;
  load_a_frag(A, arow, rowOK, K, 0, khalf, af);

  int kb = 0;
  while (true) {
    // ---- issue both B-fragment load groups up front (full MLP) -----------
    BF16Frag b0[4], b1[4];
#pragma unroll
    for (int t = 0; t < 4; ++t) {
      size_t fb = (((size_t)(tBase + t) * numKb + kb) * 32 + lane) * 2;
      b0[t].q[0] = Bq[fb + 0];
      b0[t].q[1] = Bq[fb + 1];
    }
#pragma unroll
    for (int t = 0; t < 4; ++t) {
      size_t fb = (((size_t)(tBase + 4 + t) * numKb + kb) * 32 + lane) * 2;
      b1[t].q[0] = Bq[fb + 0];
      b1[t].q[1] = Bq[fb + 1];
    }
    const bool more = (kb + 1 < numKb);

    // ---- WMMA group 0 (tiles 0..3), independent D/A/B -> no hazard NOPs --
#pragma unroll
    for (int t = 0; t < 4; ++t) {
      acc[t] = __builtin_amdgcn_wmma_f32_16x16x32_bf16(
          false, af.v, false, b0[t].v, (short)0, acc[t], false, false);
    }
    // ---- stage next A fragment while group-1 loads land ------------------
    BF16Frag afn;
    if (more) load_a_frag(A, arow, rowOK, K, kb + 1, khalf, afn);
    // ---- WMMA group 1 (tiles 4..7) ---------------------------------------
#pragma unroll
    for (int t = 0; t < 4; ++t) {
      acc[4 + t] = __builtin_amdgcn_wmma_f32_16x16x32_bf16(
          false, af.v, false, b1[t].v, (short)0, acc[4 + t], false, false);
    }
    if (!more) break;
    af = afn;
    ++kb;
  }

  // -------- epilogue: C/D f32 layout: VGPR r -> (M = r + khalf*8, N = lane&15)
#pragma unroll
  for (int t = 0; t < 8; ++t) {
    int col = (tBase + t) * 16 + (lane & 15);
#pragma unroll
    for (int r = 0; r < 8; ++r) {
      int orow = mBase + r + khalf * 8;
      if (orow < M) C[(size_t)orow * Nc + col] = acc[t][r];
    }
  }
}

// ---------------------------------------------------------------------------
// Per-node attention coefficients: al_s[n,h] = <h[n,h,:], a_src[h,:]>, same a_dst
// ---------------------------------------------------------------------------
__global__ void k_attn(const float* __restrict__ h,
                       const float* __restrict__ as_, const float* __restrict__ ad_,
                       float* __restrict__ als, float* __restrict__ ald, int n) {
  int idx = blockIdx.x * blockDim.x + threadIdx.x;
  if (idx >= n * NH) return;
  int node = idx >> 2, hd = idx & 3;
  const float* hp = h + (size_t)node * HCDIM + hd * CC;
  const float* ap = as_ + hd * CC;
  const float* bp = ad_ + hd * CC;
  float s = 0.f, d = 0.f;
#pragma unroll 8
  for (int c = 0; c < CC; ++c) { float v = hp[c]; s += v * ap[c]; d += v * bp[c]; }
  als[idx] = s; ald[idx] = d;
}

// ---------------------------------------------------------------------------
// Edge pass 1: segment max of leaky_relu(al_s[src]+al_d[dst]) over dst
// ---------------------------------------------------------------------------
__global__ void k_edge_max(const int* __restrict__ ei,
                           const float* __restrict__ als, const float* __restrict__ ald,
                           float* __restrict__ maxb) {
  size_t idx = (size_t)blockIdx.x * blockDim.x + threadIdx.x;
  if (idx >= (size_t)NEL * NH) return;
  int e = (int)(idx >> 2), hd = (int)(idx & 3);
  int s, d;
  if (e < NE) { s = ei[e]; d = ei[NE + e]; } else { s = d = e - NE; }
  float l = als[s * NH + hd] + ald[d * NH + hd];
  l = (l > 0.f) ? l : 0.2f * l;
  atomicMaxF(&maxb[d * NH + hd], l);
}

// ---------------------------------------------------------------------------
// Edge pass 2: numerator exp + segment sum
// ---------------------------------------------------------------------------
__global__ void k_edge_exp(const int* __restrict__ ei,
                           const float* __restrict__ als, const float* __restrict__ ald,
                           const float* __restrict__ maxb,
                           float* __restrict__ expb, float* __restrict__ sumb) {
  size_t idx = (size_t)blockIdx.x * blockDim.x + threadIdx.x;
  if (idx >= (size_t)NEL * NH) return;
  int e = (int)(idx >> 2), hd = (int)(idx & 3);
  int s, d;
  if (e < NE) { s = ei[e]; d = ei[NE + e]; } else { s = d = e - NE; }
  float l = als[s * NH + hd] + ald[d * NH + hd];
  l = (l > 0.f) ? l : 0.2f * l;
  float el = __expf(l - maxb[d * NH + hd]);
  expb[idx] = el;
  atomicAdd(&sumb[d * NH + hd], el);
}

// ---------------------------------------------------------------------------
// Edge pass 3: out[dst] += h[src] * alpha   (4 channels per thread, float4 gather)
// ---------------------------------------------------------------------------
__global__ void k_edge_scatter(const int* __restrict__ ei,
                               const float* __restrict__ expb, const float* __restrict__ sumb,
                               const float* __restrict__ feat, float* __restrict__ agg) {
  size_t idx = (size_t)blockIdx.x * blockDim.x + threadIdx.x;
  if (idx >= (size_t)NEL * 64) return;
  int e = (int)(idx >> 6), cg = (int)(idx & 63);
  int s, d;
  if (e < NE) { s = ei[e]; d = ei[NE + e]; } else { s = d = e - NE; }
  int hd = cg >> 4;
  float alpha = expb[(size_t)e * NH + hd] / (sumb[d * NH + hd] + 1e-16f);
  float4 hv = *reinterpret_cast<const float4*>(feat + (size_t)s * HCDIM + cg * 4);
  float* o = agg + (size_t)d * HCDIM + cg * 4;
  atomicAdd(o + 0, hv.x * alpha);
  atomicAdd(o + 1, hv.y * alpha);
  atomicAdd(o + 2, hv.z * alpha);
  atomicAdd(o + 3, hv.w * alpha);
}

// ---------------------------------------------------------------------------
// bias + LayerNorm(256) + ReLU, in place; one wave32 per node, 8 ch/lane
// ---------------------------------------------------------------------------
__global__ void k_bias_ln_relu(float* __restrict__ x,
                               const float* __restrict__ bias,
                               const float* __restrict__ gamma,
                               const float* __restrict__ beta, int n) {
  int node = (int)(((size_t)blockIdx.x * blockDim.x + threadIdx.x) >> 5);
  int lane = threadIdx.x & 31;
  if (node >= n) return;
  float* row = x + (size_t)node * HCDIM;
  float v[8], s = 0.f, sq = 0.f;
#pragma unroll
  for (int j = 0; j < 8; ++j) {
    int c = j * 32 + lane;
    v[j] = row[c] + bias[c];
    s += v[j]; sq += v[j] * v[j];
  }
#pragma unroll
  for (int o = 16; o > 0; o >>= 1) {
    s  += __shfl_xor(s, o, 32);
    sq += __shfl_xor(sq, o, 32);
  }
  float mu  = s * (1.f / HCDIM);
  float var = sq * (1.f / HCDIM) - mu * mu;
  float inv = rsqrtf(var + 1e-5f);
#pragma unroll
  for (int j = 0; j < 8; ++j) {
    int c = j * 32 + lane;
    float y = (v[j] - mu) * inv * gamma[c] + beta[c];
    row[c] = (y > 0.f) ? y : 0.f;
  }
}

// ---------------------------------------------------------------------------
// gate[n] = relu(T[n,:]+pb1) . pW2 + pb2  ; one wave32 per node (4 ch/lane)
// ---------------------------------------------------------------------------
__global__ void k_gate(const float* __restrict__ T, const float* __restrict__ pb1,
                       const float* __restrict__ pW2, const float* __restrict__ pb2,
                       float* __restrict__ gate, int n) {
  int node = (int)(((size_t)blockIdx.x * blockDim.x + threadIdx.x) >> 5);
  int lane = threadIdx.x & 31;
  if (node >= n) return;
  const float* row = T + (size_t)node * PDIM;
  float acc = 0.f;
#pragma unroll
  for (int j = 0; j < 4; ++j) {
    int c = j * 32 + lane;
    float t = row[c] + pb1[c];
    t = (t > 0.f) ? t : 0.f;
    acc += t * pW2[c];
  }
#pragma unroll
  for (int o = 16; o > 0; o >>= 1) acc += __shfl_xor(acc, o, 32);
  if (lane == 0) gate[node] = acc + pb2[0];
}

__global__ void k_gmax(const int* __restrict__ batch, const float* __restrict__ gate,
                       float* __restrict__ maxg, int n) {
  int i = blockIdx.x * blockDim.x + threadIdx.x;
  if (i < n) atomicMaxF(&maxg[batch[i]], gate[i]);
}

__global__ void k_gsum(const int* __restrict__ batch, const float* __restrict__ gate,
                       const float* __restrict__ maxg, float* __restrict__ wts,
                       float* __restrict__ sumg, int n) {
  int i = blockIdx.x * blockDim.x + threadIdx.x;
  if (i >= n) return;
  int b = batch[i];
  float e = __expf(gate[i] - maxg[b]);
  wts[i] = e;
  atomicAdd(&sumg[b], e);
}

__global__ void k_pool(const int* __restrict__ batch, const float* __restrict__ wts,
                       const float* __restrict__ sumg, const float* __restrict__ feat,
                       float* __restrict__ pooled) {
  size_t idx = (size_t)blockIdx.x * blockDim.x + threadIdx.x;
  if (idx >= (size_t)NN * 64) return;
  int node = (int)(idx >> 6), cg = (int)(idx & 63);
  int b = batch[node];
  float w = wts[node] / (sumg[b] + 1e-16f);
  float4 hv = *reinterpret_cast<const float4*>(feat + (size_t)node * HCDIM + cg * 4);
  float* o = pooled + (size_t)b * HCDIM + cg * 4;
  atomicAdd(o + 0, hv.x * w);
  atomicAdd(o + 1, hv.y * w);
  atomicAdd(o + 2, hv.z * w);
  atomicAdd(o + 3, hv.w * w);
}

// ---------------------------------------------------------------------------
// classifier: out[g,:2] = relu(pooled[g] @ cW1 + cb1) @ cW2 + cb2
// ---------------------------------------------------------------------------
__global__ void k_classifier(const float* __restrict__ pooled,
                             const float* __restrict__ cW1, const float* __restrict__ cb1,
                             const float* __restrict__ cW2, const float* __restrict__ cb2,
                             float* __restrict__ out) {
  __shared__ float hid[PDIM];
  int g = blockIdx.x, t = threadIdx.x;
  const float* p = pooled + (size_t)g * HCDIM;
  float a = cb1[t];
  for (int k = 0; k < HCDIM; ++k) a += p[k] * cW1[(size_t)k * PDIM + t];
  hid[t] = (a > 0.f) ? a : 0.f;
  __syncthreads();
  if (t < 2) {
    float o = cb2[t];
    for (int k = 0; k < PDIM; ++k) o += hid[k] * cW2[k * 2 + t];
    out[g * 2 + t] = o;
  }
}

// ---------------------------------------------------------------------------
// Host-side orchestration (graph-capture safe: only launches on `stream`)
// ---------------------------------------------------------------------------
extern "C" void kernel_launch(void* const* d_in, const int* in_sizes, int n_in,
                              void* d_out, int out_size, void* d_ws, size_t ws_size,
                              hipStream_t stream) {
  const float* x    = (const float*)d_in[0];
  const int*   ei   = (const int*)d_in[1];
  const int*   batc = (const int*)d_in[2];
  const float* W0   = (const float*)d_in[3];
  const float* as0  = (const float*)d_in[4];
  const float* ad0  = (const float*)d_in[5];
  const float* b0   = (const float*)d_in[6];
  const float* g0   = (const float*)d_in[7];
  const float* be0  = (const float*)d_in[8];
  const float* W1   = (const float*)d_in[9];
  const float* as1  = (const float*)d_in[10];
  const float* ad1  = (const float*)d_in[11];
  const float* b1   = (const float*)d_in[12];
  const float* g1   = (const float*)d_in[13];
  const float* be1  = (const float*)d_in[14];
  const float* pW1  = (const float*)d_in[15];
  const float* pb1  = (const float*)d_in[16];
  const float* pW2  = (const float*)d_in[17];
  const float* pb2  = (const float*)d_in[18];
  const float* cW1  = (const float*)d_in[19];
  const float* cb1  = (const float*)d_in[20];
  const float* cW2  = (const float*)d_in[21];
  const float* cb2  = (const float*)d_in[22];
  float* out = (float*)d_out;

  char* ws = (char*)d_ws;
  size_t off = 0;
  auto alloc = [&](size_t bytes) -> void* {
    void* p = ws + off;
    off += (bytes + 255) & ~(size_t)255;
    return p;
  };
  float* hA   = (float*)alloc((size_t)NN * HCDIM * 4);   // GEMM out / pre-agg h
  float* hB   = (float*)alloc((size_t)NN * HCDIM * 4);   // aggregated / LN'd features
  float* als  = (float*)alloc((size_t)NN * NH * 4);
  float* ald  = (float*)alloc((size_t)NN * NH * 4);
  float* maxb = (float*)alloc((size_t)NN * NH * 4);
  float* sumb = (float*)alloc((size_t)NN * NH * 4);
  float* expb = (float*)alloc((size_t)NEL * NH * 4);
  float* gate = (float*)alloc((size_t)NN * 4);
  float* wts  = (float*)alloc((size_t)NN * 4);
  float* maxg = (float*)alloc((size_t)NG * 4);
  float* sumg = (float*)alloc((size_t)NG * 4);
  float* pool = (float*)alloc((size_t)NG * HCDIM * 4);
  unsigned short* W0f  = (unsigned short*)alloc((size_t)FIN   * HCDIM * 2);
  unsigned short* W1f  = (unsigned short*)alloc((size_t)HCDIM * HCDIM * 2);
  unsigned short* pW1f = (unsigned short*)alloc((size_t)HCDIM * PDIM  * 2);

  const int T = 256;
  const int gemmBlksM = (NN + 127) / 128;
  const size_t elh = (size_t)NEL * NH;
  const size_t elc = (size_t)NEL * 64;
  auto blks = [&](size_t n) { return (unsigned)((n + T - 1) / T); };

  // ---- weight conversion (fragment-major bf16) ----
  k_convert_w<<<blks(FIN * HCDIM),   T, 0, stream>>>(W0,  W0f,  FIN,   HCDIM);
  k_convert_w<<<blks(HCDIM * HCDIM), T, 0, stream>>>(W1,  W1f,  HCDIM, HCDIM);
  k_convert_w<<<blks(HCDIM * PDIM),  T, 0, stream>>>(pW1, pW1f, HCDIM, PDIM);

  // ==================== GAT layer 0 ====================
  k_gemm_bf16<<<dim3(gemmBlksM, 2), T, 0, stream>>>(x, W0f, hA, NN, FIN, HCDIM);
  k_attn<<<blks((size_t)NN * NH), T, 0, stream>>>(hA, as0, ad0, als, ald, NN);
  k_fill<<<blks((size_t)NN * NH), T, 0, stream>>>(maxb, -1e30f, (size_t)NN * NH);
  k_fill<<<blks((size_t)NN * NH), T, 0, stream>>>(sumb, 0.f,    (size_t)NN * NH);
  k_fill<<<blks((size_t)NN * HCDIM), T, 0, stream>>>(hB, 0.f, (size_t)NN * HCDIM);
  k_edge_max<<<blks(elh), T, 0, stream>>>(ei, als, ald, maxb);
  k_edge_exp<<<blks(elh), T, 0, stream>>>(ei, als, ald, maxb, expb, sumb);
  k_edge_scatter<<<blks(elc), T, 0, stream>>>(ei, expb, sumb, hA, hB);
  k_bias_ln_relu<<<blks((size_t)NN * 32), T, 0, stream>>>(hB, b0, g0, be0, NN);

  // ==================== GAT layer 1 ====================
  k_gemm_bf16<<<dim3(gemmBlksM, 2), T, 0, stream>>>(hB, W1f, hA, NN, HCDIM, HCDIM);
  k_attn<<<blks((size_t)NN * NH), T, 0, stream>>>(hA, as1, ad1, als, ald, NN);
  k_fill<<<blks((size_t)NN * NH), T, 0, stream>>>(maxb, -1e30f, (size_t)NN * NH);
  k_fill<<<blks((size_t)NN * NH), T, 0, stream>>>(sumb, 0.f,    (size_t)NN * NH);
  k_fill<<<blks((size_t)NN * HCDIM), T, 0, stream>>>(hB, 0.f, (size_t)NN * HCDIM);
  k_edge_max<<<blks(elh), T, 0, stream>>>(ei, als, ald, maxb);
  k_edge_exp<<<blks(elh), T, 0, stream>>>(ei, als, ald, maxb, expb, sumb);
  k_edge_scatter<<<blks(elc), T, 0, stream>>>(ei, expb, sumb, hA, hB);
  k_bias_ln_relu<<<blks((size_t)NN * 32), T, 0, stream>>>(hB, b1, g1, be1, NN);

  // ==================== global attention pooling ====================
  k_gemm_bf16<<<dim3(gemmBlksM, 1), T, 0, stream>>>(hB, pW1f, hA, NN, HCDIM, PDIM);
  k_gate<<<blks((size_t)NN * 32), T, 0, stream>>>(hA, pb1, pW2, pb2, gate, NN);
  k_fill<<<1, NG, 0, stream>>>(maxg, -1e30f, (size_t)NG);
  k_fill<<<1, NG, 0, stream>>>(sumg, 0.f, (size_t)NG);
  k_fill<<<blks((size_t)NG * HCDIM), T, 0, stream>>>(pool, 0.f, (size_t)NG * HCDIM);
  k_gmax<<<blks(NN), T, 0, stream>>>(batc, gate, maxg, NN);
  k_gsum<<<blks(NN), T, 0, stream>>>(batc, gate, maxg, wts, sumg, NN);
  k_pool<<<blks((size_t)NN * 64), T, 0, stream>>>(batc, wts, sumg, hB, pool);

  // ==================== classifier ====================
  k_classifier<<<NG, PDIM, 0, stream>>>(pool, cW1, cb1, cW2, cb2, out);
}
